// ADEA_41927470744109
// MI455X (gfx1250) — matrix operand order repaired and checked by reading, not compiled, not run on hardware
//
#include <hip/hip_runtime.h>
#include <cstdint>
#include <cstddef>

// ---------------- problem constants (from the reference) ----------------
namespace {
constexpr int kN  = 50000;     // entities
constexpr int kD  = 256;       // entity dim
constexpr int kH  = 4;         // heads
constexpr int kDH = 64;        // dim per head
constexpr int kE  = 200000;    // entity-entity edges
constexpr int kER = 150000;    // entity-relation edges
constexpr int kEA = 150000;    // entity-attribute edges
constexpr int kL  = 2;         // layers
constexpr int kEntCols  = kL * kD + 2 * kDH;   // 640
constexpr int kConcCols = 2 * kDH;             // 128
constexpr float kNegBig = -3.402823466e38f;
}
static_assert(kN % 16 == 0, "node tiles must cover N exactly");

typedef __attribute__((ext_vector_type(16))) __bf16 v16bf;
typedef __attribute__((ext_vector_type(8)))  float  v8f;

// ---------------- small helpers ----------------
__device__ inline void atomicMaxF(float* addr, float val) {
    unsigned int* ua = (unsigned int*)addr;
    unsigned int old = *ua;
    while (true) {
        float cur = __uint_as_float(old);
        if (cur >= val) break;
        unsigned int assumed = old;
        old = atomicCAS(ua, assumed, __float_as_uint(val));
        if (old == assumed) break;
    }
}

__global__ void fill_f32(float* __restrict__ p, float v, int n) {
    int i = blockIdx.x * blockDim.x + threadIdx.x;
    if (i < n) p[i] = v;
}

// ---------------- concept (mean + relu) aggregation ----------------
// w = seg_softmax(ones) == 1/deg(row)  ->  mean over incoming embeddings.
// Sums accumulate directly into the `conc` region of d_out (stride 128).
__global__ void concept_accum(const float* __restrict__ emb,
                              const int* __restrict__ idx,
                              float* __restrict__ sumBase,  // d_out + N*640 + which*64
                              float* __restrict__ cnt,
                              int nedges) {
    int t = blockIdx.x * blockDim.x + threadIdx.x;
    int e = t >> 6;
    int d = t & 63;
    if (e >= nedges) return;
    int r = idx[2 * e];
    int c = idx[2 * e + 1];
    atomicAdd(&sumBase[(size_t)r * kConcCols + d], emb[(size_t)c * 64 + d]);
    if (d == 0) atomicAdd(&cnt[r], 1.0f);
}

__global__ void concept_final(float* __restrict__ out,
                              const float* __restrict__ cnt,
                              int which) {
    int t = blockIdx.x * blockDim.x + threadIdx.x;
    if (t >= kN * 64) return;
    int n = t >> 6;
    int d = t & 63;
    size_t cbase = (size_t)kN * kEntCols + (size_t)n * kConcCols + which * 64 + d;
    float cv = cnt[n];
    float v = (cv > 0.f) ? fmaxf(out[cbase] / cv, 0.f) : 0.f;
    out[cbase] = v;                                                  // conc output
    out[(size_t)n * kEntCols + kL * kD + which * 64 + d] = v;        // ent_out tail
}

// ---------------- initial h = segsum(softmax(edge_val) * ent_emb[col]) ----------------
__global__ void ev_rowmax(const float* __restrict__ ev, const int* __restrict__ eidx,
                          float* __restrict__ rowMaxV) {
    int e = blockIdx.x * blockDim.x + threadIdx.x;
    if (e >= kE) return;
    atomicMaxF(&rowMaxV[eidx[2 * e]], ev[e]);
}

__global__ void ev_expsum(const float* __restrict__ ev, const int* __restrict__ eidx,
                          const float* __restrict__ rowMaxV,
                          float* __restrict__ evw, float* __restrict__ rowSumV) {
    int e = blockIdx.x * blockDim.x + threadIdx.x;
    if (e >= kE) return;
    int r = eidx[2 * e];
    float x = expf(ev[e] - rowMaxV[r]);
    evw[e] = x;
    atomicAdd(&rowSumV[r], x);
}

__global__ void init_h_aggregate(const float* __restrict__ ent_emb,
                                 const float* __restrict__ evw,
                                 const float* __restrict__ rowSumV,
                                 const int* __restrict__ eidx,
                                 float* __restrict__ h) {
    int e = blockIdx.x;      // one edge per block
    int d = threadIdx.x;     // 256 features
    int r = eidx[2 * e];
    int c = eidx[2 * e + 1];
    float w = evw[e] / rowSumV[r];
    atomicAdd(&h[(size_t)r * kD + d], w * ent_emb[(size_t)c * kD + d]);
}

// ---------------- WMMA node transform: S1/S2[n,h] = relu(relu(h)@Wk_h) . ak ----------------
// One wave (32 threads) per (16-node tile, head). A = relu(h) tile (16x64, bf16),
// B = Wk_h (64x64, bf16), fp32 accumulate, two chained 16x16x32 WMMAs per K=64.
__global__ __launch_bounds__(32)
void node_transform_wmma(const float* __restrict__ h,
                         const float* __restrict__ Wk,   // [H][64][64] (d,f) for this layer
                         const float* __restrict__ ak,   // [H][128]    for this layer
                         float* __restrict__ S1,
                         float* __restrict__ S2) {
    const int lane = threadIdx.x;       // 0..31
    const int tile = blockIdx.x;        // node tile
    const int head = blockIdx.y;        // 0..3
    const int half = lane >> 4;         // 0/1
    const int l15  = lane & 15;
    const int mrow = tile * 16 + l15;

    const float* hrow = h + (size_t)mrow * kD + head * kDH;

    // A fragments (16x32 bf16): lane group g, VGPR v holds K = 16*(v>>2) + 8*g + 2*(v&3) + {0,1}
    v16bf A0, A1;
#pragma unroll
    for (int v = 0; v < 8; ++v) {
        int k = ((v >> 2) << 4) + (half << 3) + ((v & 3) << 1);
        float2 f0 = *(const float2*)(hrow + k);        // K-half 0..31
        float2 f1 = *(const float2*)(hrow + 32 + k);   // K-half 32..63
        A0[2 * v]     = (__bf16)fmaxf(f0.x, 0.f);
        A0[2 * v + 1] = (__bf16)fmaxf(f0.y, 0.f);
        A1[2 * v]     = (__bf16)fmaxf(f1.x, 0.f);
        A1[2 * v + 1] = (__bf16)fmaxf(f1.y, 0.f);
    }

    const float* W  = Wk + (size_t)head * kDH * kDH;
    const float* a1 = ak + (size_t)head * (2 * kDH);
    const float* a2 = a1 + kDH;

    float p1[8], p2[8];
#pragma unroll
    for (int v = 0; v < 8; ++v) { p1[v] = 0.f; p2[v] = 0.f; }

#pragma unroll
    for (int t = 0; t < 4; ++t) {               // 4 N-tiles of 16 outputs
        int f = t * 16 + l15;
        // B fragments (32x16 bf16): lane group g, VGPR v holds K = 16*(v>>2) + 8*g + 2*(v&3) + {0,1}
        v16bf B0, B1;
#pragma unroll
        for (int v = 0; v < 8; ++v) {
            int k = ((v >> 2) << 4) + (half << 3) + ((v & 3) << 1);
            B0[2 * v]     = (__bf16)W[(size_t)(k)     * kDH + f];
            B0[2 * v + 1] = (__bf16)W[(size_t)(k + 1) * kDH + f];
            B1[2 * v]     = (__bf16)W[(size_t)(32 + k)     * kDH + f];
            B1[2 * v + 1] = (__bf16)W[(size_t)(32 + k + 1) * kDH + f];
        }
        v8f c = {};
        c = __builtin_amdgcn_wmma_f32_16x16x32_bf16(false, A0, false, B0,
                                                    (short)0, c, false, false);
        c = __builtin_amdgcn_wmma_f32_16x16x32_bf16(false, A1, false, B1,
                                                    (short)0, c, false, false);
        float w1 = a1[f], w2 = a2[f];
#pragma unroll
        for (int v = 0; v < 8; ++v) {           // D: lane holds (M = v + 8*half, N = f)
            float r = fmaxf(c[v], 0.f);
            p1[v] += r * w1;
            p2[v] += r * w2;
        }
    }

    // reduce over the 16 N-lanes within each half-wave
#pragma unroll
    for (int off = 1; off < 16; off <<= 1) {
#pragma unroll
        for (int v = 0; v < 8; ++v) {
            p1[v] += __shfl_xor(p1[v], off, 32);
            p2[v] += __shfl_xor(p2[v], off, 32);
        }
    }
    if (l15 == 0) {
#pragma unroll
        for (int v = 0; v < 8; ++v) {
            int m = tile * 16 + v + 8 * half;
            S1[(size_t)m * kH + head] = p1[v];
            S2[(size_t)m * kH + head] = p2[v];
        }
    }
}

// ---------------- edge attention passes ----------------
__global__ void edge_logits(const float* __restrict__ S1, const float* __restrict__ S2,
                            const int* __restrict__ eidx,
                            float* __restrict__ attnbuf, float* __restrict__ headMax) {
    __shared__ float red[256];
    int e = blockIdx.x * 256 + threadIdx.x;
    float lm[kH];
#pragma unroll
    for (int h2 = 0; h2 < kH; ++h2) lm[h2] = kNegBig;
    if (e < kE) {
        int r = eidx[2 * e];
        int c = eidx[2 * e + 1];
#pragma unroll
        for (int h2 = 0; h2 < kH; ++h2) {
            float a = S1[(size_t)r * kH + h2] + S2[(size_t)c * kH + h2];
            a = (a > 0.f) ? a : 0.3f * a;      // leaky_relu(0.3)
            attnbuf[(size_t)e * kH + h2] = a;
            lm[h2] = a;
        }
    }
#pragma unroll
    for (int h2 = 0; h2 < kH; ++h2) {
        red[threadIdx.x] = lm[h2];
        __syncthreads();
        for (int s = 128; s > 0; s >>= 1) {
            if (threadIdx.x < s) red[threadIdx.x] = fmaxf(red[threadIdx.x], red[threadIdx.x + s]);
            __syncthreads();
        }
        if (threadIdx.x == 0) atomicMaxF(&headMax[h2], red[0]);
        __syncthreads();
    }
}

__global__ void edge_headsum_rowmax(const float* __restrict__ attnbuf,
                                    const int* __restrict__ eidx,
                                    const float* __restrict__ headMax,
                                    float* __restrict__ headSum,
                                    float* __restrict__ rowMaxA) {
    __shared__ float red[256];
    int e = blockIdx.x * 256 + threadIdx.x;
    float ls[kH] = {0.f, 0.f, 0.f, 0.f};
    if (e < kE) {
        int r = eidx[2 * e];
#pragma unroll
        for (int h2 = 0; h2 < kH; ++h2) {
            float a = attnbuf[(size_t)e * kH + h2];
            ls[h2] = expf(a - headMax[h2]);
            atomicMaxF(&rowMaxA[(size_t)r * kH + h2], a);
        }
    }
#pragma unroll
    for (int h2 = 0; h2 < kH; ++h2) {
        red[threadIdx.x] = ls[h2];
        __syncthreads();
        for (int s = 128; s > 0; s >>= 1) {
            if (threadIdx.x < s) red[threadIdx.x] += red[threadIdx.x + s];
            __syncthreads();
        }
        if (threadIdx.x == 0) atomicAdd(&headSum[h2], red[0]);
        __syncthreads();
    }
}

// g = exp(a - M)/S (global softmax over E), then e2 = exp(g - segmax(g)) with
// segmax(g) = exp(segmax(a) - M)/S (monotonicity). Overwrites attnbuf with e2.
__global__ void edge_segexp(float* __restrict__ attnbuf, const int* __restrict__ eidx,
                            const float* __restrict__ headMax, const float* __restrict__ headSum,
                            const float* __restrict__ rowMaxA, float* __restrict__ rowSum) {
    int t = blockIdx.x * blockDim.x + threadIdx.x;
    int e = t >> 2;
    int h2 = t & 3;
    if (e >= kE) return;
    int r = eidx[2 * e];
    float M = headMax[h2];
    float invS = 1.f / headSum[h2];
    float g  = expf(attnbuf[(size_t)e * kH + h2] - M) * invS;
    float gm = expf(rowMaxA[(size_t)r * kH + h2] - M) * invS;
    float e2 = expf(g - gm);
    attnbuf[(size_t)e * kH + h2] = e2;
    atomicAdd(&rowSum[(size_t)r * kH + h2], e2);
}

__global__ void edge_aggregate(const float* __restrict__ h,
                               const float* __restrict__ attnbuf,
                               const float* __restrict__ rowSum,
                               const int* __restrict__ eidx,
                               float* __restrict__ hnew) {
    int e = blockIdx.x;      // one edge per block
    int d = threadIdx.x;     // 256 features
    int r = eidx[2 * e];
    int c = eidx[2 * e + 1];
    int hd = d >> 6;
    float w = attnbuf[(size_t)e * kH + hd] / rowSum[(size_t)r * kH + hd];
    float v = w * fmaxf(h[(size_t)c * kD + d], 0.f);   // nb = relu(h)[col]
    atomicAdd(&hnew[(size_t)r * kD + d], v);
}

__global__ void layer_final(const float* __restrict__ hnew,
                            float* __restrict__ h,
                            float* __restrict__ out, int layer) {
    int t = blockIdx.x * blockDim.x + threadIdx.x;
    if (t >= kN * kD) return;
    int n = t / kD;
    int d = t - n * kD;
    float x = tanhf(hnew[t]);
    h[t] = x;                                              // feeds next layer
    out[(size_t)n * kEntCols + layer * kD + d] = x;        // outs[layer] slot
}

// ---------------- driver ----------------
extern "C" void kernel_launch(void* const* d_in, const int* in_sizes, int n_in,
                              void* d_out, int out_size, void* d_ws, size_t ws_size,
                              hipStream_t stream) {
    (void)in_sizes; (void)n_in; (void)out_size; (void)ws_size;

    const float* ent_emb  = (const float*)d_in[0];
    const float* rel_emb  = (const float*)d_in[1];
    const float* attr_emb = (const float*)d_in[2];
    const float* ent_k    = (const float*)d_in[3];   // [L][H][64][64]
    const float* ent_ak   = (const float*)d_in[4];   // [L][H][128]
    const float* edge_val = (const float*)d_in[5];
    const int*   eidx     = (const int*)d_in[6];
    const int*   ridx     = (const int*)d_in[7];
    const int*   aidx     = (const int*)d_in[8];
    float* out = (float*)d_out;

    // workspace layout (floats)
    float* W = (float*)d_ws;
    size_t o = 0;
    float* h       = W + o; o += (size_t)kN * kD;
    float* hnew    = W + o; o += (size_t)kN * kD;
    float* S1      = W + o; o += (size_t)kN * kH;
    float* S2      = W + o; o += (size_t)kN * kH;
    float* attnbuf = W + o; o += (size_t)kE * kH;
    float* rowMaxA = W + o; o += (size_t)kN * kH;
    float* rowSum  = W + o; o += (size_t)kN * kH;
    float* headMax = W + o; o += 64;
    float* headSum = W + o; o += 64;
    float* evw     = W + o; o += (size_t)kE;
    float* rowMaxV = W + o; o += (size_t)kN;
    float* rowSumV = W + o; o += (size_t)kN;
    float* cntR    = W + o; o += (size_t)kN;
    float* cntA    = W + o; o += (size_t)kN;

    const int TB = 256;
    auto blocks = [](long long n, int tb) { return (unsigned)((n + tb - 1) / tb); };

    // ---- concepts (independent of the GNN chain, same stream) ----
    hipMemsetAsync(out + (size_t)kN * kEntCols, 0, (size_t)kN * kConcCols * sizeof(float), stream);
    hipMemsetAsync(cntR, 0, (size_t)kN * sizeof(float), stream);
    hipMemsetAsync(cntA, 0, (size_t)kN * sizeof(float), stream);
    concept_accum<<<blocks((long long)kER * 64, TB), TB, 0, stream>>>(
        rel_emb, ridx, out + (size_t)kN * kEntCols + 0, cntR, kER);
    concept_accum<<<blocks((long long)kEA * 64, TB), TB, 0, stream>>>(
        attr_emb, aidx, out + (size_t)kN * kEntCols + 64, cntA, kEA);
    concept_final<<<blocks((long long)kN * 64, TB), TB, 0, stream>>>(out, cntR, 0);
    concept_final<<<blocks((long long)kN * 64, TB), TB, 0, stream>>>(out, cntA, 1);

    // ---- initial h: softmax(edge_val) weighted aggregation of ent_emb ----
    fill_f32<<<blocks(kN, TB), TB, 0, stream>>>(rowMaxV, kNegBig, kN);
    hipMemsetAsync(rowSumV, 0, (size_t)kN * sizeof(float), stream);
    hipMemsetAsync(h, 0, (size_t)kN * kD * sizeof(float), stream);
    ev_rowmax<<<blocks(kE, TB), TB, 0, stream>>>(edge_val, eidx, rowMaxV);
    ev_expsum<<<blocks(kE, TB), TB, 0, stream>>>(edge_val, eidx, rowMaxV, evw, rowSumV);
    init_h_aggregate<<<kE, kD, 0, stream>>>(ent_emb, evw, rowSumV, eidx, h);

    // ---- GAT layers ----
    for (int l = 0; l < kL; ++l) {
        const float* Wl  = ent_k  + (size_t)l * kH * kDH * kDH;
        const float* akl = ent_ak + (size_t)l * kH * 2 * kDH;

        node_transform_wmma<<<dim3(kN / 16, kH), 32, 0, stream>>>(h, Wl, akl, S1, S2);

        fill_f32<<<1, 64, 0, stream>>>(headMax, kNegBig, kH);
        hipMemsetAsync(headSum, 0, kH * sizeof(float), stream);
        fill_f32<<<blocks((long long)kN * kH, TB), TB, 0, stream>>>(rowMaxA, kNegBig, kN * kH);
        hipMemsetAsync(rowSum, 0, (size_t)kN * kH * sizeof(float), stream);
        hipMemsetAsync(hnew, 0, (size_t)kN * kD * sizeof(float), stream);

        edge_logits<<<blocks(kE, TB), TB, 0, stream>>>(S1, S2, eidx, attnbuf, headMax);
        edge_headsum_rowmax<<<blocks(kE, TB), TB, 0, stream>>>(attnbuf, eidx, headMax,
                                                               headSum, rowMaxA);
        edge_segexp<<<blocks((long long)kE * kH, TB), TB, 0, stream>>>(attnbuf, eidx, headMax,
                                                                       headSum, rowMaxA, rowSum);
        edge_aggregate<<<kE, kD, 0, stream>>>(h, attnbuf, rowSum, eidx, hnew);
        layer_final<<<blocks((long long)kN * kD, TB), TB, 0, stream>>>(hnew, h, out, l);
    }
}
